// PolyLang_60919816126774
// MI455X (gfx1250) — matrix operand order repaired and checked by reading, not compiled, not run on hardware
//
#include <hip/hip_runtime.h>
#include <math.h>

// ---- model constants ----
#define LNUM 12
#define HNUM 12
#define CDIM 768
#define VDIM 50257
#define BDIM 4
#define TDIM 1024
#define HD   64

// ---- WMMA types (CDNA5 wave32, v_wmma_f32_16x16x32_bf16) ----
typedef __attribute__((ext_vector_type(16))) __bf16 v16bf;
typedef __attribute__((ext_vector_type(8)))  __bf16 bf16x8;
typedef __attribute__((ext_vector_type(8)))  float  v8f;

union FragA { v16bf v; bf16x8 h[2]; };

__device__ __forceinline__ v8f wmma_bf16(const FragA& a, const FragA& b, v8f c) {
    return __builtin_amdgcn_wmma_f32_16x16x32_bf16(
        /*neg_a=*/false, a.v, /*neg_b=*/false, b.v,
        /*c_mod=*/(short)0, c, /*reuse_a=*/false, /*reuse_b=*/false);
}

// =====================================================================
// Embedding: x[b,t,:] = wte[idx[b,t],:] + wpe[t,:]
// =====================================================================
__global__ __launch_bounds__(256)
void embed_kernel(const int* __restrict__ idx, const float* __restrict__ wte,
                  const float* __restrict__ wpe, float* __restrict__ x) {
    int row = blockIdx.x;              // 0..B*T-1
    int t   = row & (TDIM - 1);
    int tok = idx[row];
    const float* we = wte + (size_t)tok * CDIM;
    const float* pe = wpe + (size_t)t   * CDIM;
    float* xr = x + (size_t)row * CDIM;
    for (int c = threadIdx.x; c < CDIM; c += 256)
        xr[c] = we[c] + pe[c];
}

// =====================================================================
// LayerNorm fp32 -> bf16 (per row of CDIM)
// =====================================================================
__global__ __launch_bounds__(256)
void layernorm_bf16(const float* __restrict__ x, const float* __restrict__ g,
                    const float* __restrict__ beta, __bf16* __restrict__ out) {
    __shared__ float red[256];
    int row = blockIdx.x;
    int tid = threadIdx.x;
    const float* xr = x + (size_t)row * CDIM;

    float s = 0.f;
    for (int c = tid; c < CDIM; c += 256) s += xr[c];
    red[tid] = s; __syncthreads();
    for (int o = 128; o > 0; o >>= 1) { if (tid < o) red[tid] += red[tid + o]; __syncthreads(); }
    float mu = red[0] * (1.f / CDIM);
    __syncthreads();

    float v = 0.f;
    for (int c = tid; c < CDIM; c += 256) { float d = xr[c] - mu; v += d * d; }
    red[tid] = v; __syncthreads();
    for (int o = 128; o > 0; o >>= 1) { if (tid < o) red[tid] += red[tid + o]; __syncthreads(); }
    float rstd = rsqrtf(red[0] * (1.f / CDIM) + 1e-5f);

    __bf16* orow = out + (size_t)row * CDIM;
    for (int c = tid; c < CDIM; c += 256)
        orow[c] = (__bf16)((xr[c] - mu) * rstd * g[c] + beta[c]);
}

// =====================================================================
// WMMA GEMM: out[M,N] = A_bf16[M,K] @ W + bias (+gelu) (+residual)
//   W is fp32: row-major [K,N] (wtrans=0) or [N,K] (wtrans=1, LM head).
//   Block: 64x64 C-tile, 128 threads = 4 waves, wave = 16 rows x 64 cols.
//   BK=64: 8 WMMAs per barrier pair; B-fragments preloaded so the WMMA
//   burst issues back-to-back without per-op s_wait_dscnt stalls.
//   LDS: A tile [64][64] bf16, W tile transposed [n][k]=[64][64] bf16
//   (B-fragments become contiguous b128 reads per ISA 7.12.2 B layout).
// =====================================================================
__global__ __launch_bounds__(128)
void gemm_wmma(const __bf16* __restrict__ A, const float* __restrict__ W,
               const float* __restrict__ bias, const float* __restrict__ resid,
               float* __restrict__ outF, __bf16* __restrict__ outB,
               int M, int N, int K, int wtrans, int act) {
    __shared__ __align__(16) __bf16 At[64][64];
    __shared__ __align__(16) __bf16 Wt[64][64];   // [n][k]

    const int tid  = threadIdx.x;
    const int lane = tid & 31;
    const int wave = tid >> 5;
    const int half = lane >> 4;
    const int l16  = lane & 15;
    const int n0   = blockIdx.x * 64;
    const int m0   = blockIdx.y * 64;

    v8f acc[4] = {v8f{}, v8f{}, v8f{}, v8f{}};

    for (int kk = 0; kk < K; kk += 64) {
        // ---- A tile: 64 rows x 64 cols bf16, 32 bf16 per thread ----
        {
            int r = tid >> 1, c = (tid & 1) * 32;
            const uint4* src = (const uint4*)(A + (size_t)(m0 + r) * K + kk + c);
            uint4 u0 = src[0], u1 = src[1], u2 = src[2], u3 = src[3];
            *(uint4*)&At[r][c]      = u0;
            *(uint4*)&At[r][c + 8]  = u1;
            *(uint4*)&At[r][c + 16] = u2;
            *(uint4*)&At[r][c + 24] = u3;
            if (kk + 64 < K)
                __builtin_prefetch(A + (size_t)(m0 + r) * K + kk + 64 + c, 0, 3);
        }
        // ---- W tile -> Wt[n][k] (fp32 -> bf16 conversion on the fly) ----
        if (!wtrans) {
            int k = tid & 63, nb = (tid >> 6) * 32;
            const float4* src = (const float4*)(W + (size_t)(kk + k) * N + n0 + nb);
            #pragma unroll
            for (int i = 0; i < 8; ++i) {
                float4 f = src[i];
                Wt[nb + 4*i + 0][k] = (__bf16)f.x;
                Wt[nb + 4*i + 1][k] = (__bf16)f.y;
                Wt[nb + 4*i + 2][k] = (__bf16)f.z;
                Wt[nb + 4*i + 3][k] = (__bf16)f.w;
            }
        } else {
            int n = tid >> 1, kc = (tid & 1) * 32;
            __bf16 tmp[32];
            if (n0 + n < N) {
                const float4* src = (const float4*)(W + (size_t)(n0 + n) * K + kk + kc);
                #pragma unroll
                for (int i = 0; i < 8; ++i) {
                    float4 f = src[i];
                    tmp[4*i+0] = (__bf16)f.x; tmp[4*i+1] = (__bf16)f.y;
                    tmp[4*i+2] = (__bf16)f.z; tmp[4*i+3] = (__bf16)f.w;
                }
            } else {
                #pragma unroll
                for (int i = 0; i < 32; ++i) tmp[i] = (__bf16)0.f;
            }
            #pragma unroll
            for (int i = 0; i < 4; ++i)
                *(uint4*)&Wt[n][kc + 8*i] = *(const uint4*)&tmp[8*i];
        }
        __syncthreads();

        // ---- two 16x16x32 k-steps per tile, fragments preloaded ----
        #pragma unroll
        for (int ks = 0; ks < 64; ks += 32) {
            FragA a;
            {
                int row = wave * 16 + l16;     // A: M = lane%16
                int klo = ks + half * 16;      // lanes>=16 hold K 16..31
                a.h[0] = *(const bf16x8*)&At[row][klo];
                a.h[1] = *(const bf16x8*)&At[row][klo + 8];
            }
            FragA b[4];
            #pragma unroll
            for (int nf = 0; nf < 4; ++nf) {
                int n = nf * 16 + l16;         // B: N = lane%16, K = half*16+0..15
                b[nf].h[0] = *(const bf16x8*)&Wt[n][ks + half * 16];
                b[nf].h[1] = *(const bf16x8*)&Wt[n][ks + half * 16 + 8];
            }
            #pragma unroll
            for (int nf = 0; nf < 4; ++nf)
                acc[nf] = wmma_bf16(a, b[nf], acc[nf]);
        }
        __syncthreads();
    }

    // ---- epilogue: C/D layout: VGPR j -> row j + 8*half, col = lane%16 ----
    #pragma unroll
    for (int nf = 0; nf < 4; ++nf) {
        #pragma unroll
        for (int j = 0; j < 8; ++j) {
            int row = m0 + wave * 16 + j + 8 * half;
            int col = n0 + nf * 16 + l16;
            if (col < N) {
                float v = acc[nf][j];
                if (bias) v += bias[col];
                if (act == 1) v = 0.5f * v * (1.f + erff(v * 0.70710678118654752f));
                size_t off = (size_t)row * N + col;
                if (resid) v += resid[off];
                if (outF) outF[off] = v;
                if (outB) outB[off] = (__bf16)v;
            }
        }
    }
}

// =====================================================================
// Flash attention (NO causal mask, matching the reference).
// Block = one (b, h, 64-query tile); 128 threads = 4 waves x 16 query rows.
// S = (Q*scale) K^T via 2 WMMA k-steps; online softmax; O += P V via WMMA.
// qkv is fp32 [B*T, 3C]; output y is bf16 [B*T, C].
// =====================================================================
__global__ __launch_bounds__(128)
void attn_flash(const float* __restrict__ qkv, __bf16* __restrict__ y) {
    __shared__ __align__(16) __bf16 Kt[64][64];        // [key][d]
    __shared__ __align__(16) __bf16 Vt[64][64];        // [d][key]  (transposed)
    __shared__ __align__(16) __bf16 Ps[4][16][32];     // per-wave P scratch

    const int tid  = threadIdx.x;
    const int lane = tid & 31;
    const int wave = tid >> 5;
    const int half = lane >> 4;
    const int l16  = lane & 15;

    int blk = blockIdx.x;
    int qt  = blk & 15;                 // T/64 = 16 query tiles
    int h   = (blk >> 4) % HNUM;
    int b   = blk / (16 * HNUM);
    const float scale = 0.125f;         // 1/sqrt(64), folded into Q

    // ---- Q fragments (d 0..31 and 32..63), scaled, bf16 ----
    FragA a0, a1;
    {
        int trow = qt * 64 + wave * 16 + l16;
        const float* qp = qkv + (size_t)(b * TDIM + trow) * (3 * CDIM) + h * HD;
        int klo = half * 16;
        #pragma unroll
        for (int fr = 0; fr < 2; ++fr) {
            const float* base = qp + fr * 32;
            #pragma unroll
            for (int g = 0; g < 2; ++g) {
                const float* p = base + klo + g * 8;
                float4 f0 = *(const float4*)p;
                float4 f1 = *(const float4*)(p + 4);
                bf16x8 r;
                r[0]=(__bf16)(f0.x*scale); r[1]=(__bf16)(f0.y*scale);
                r[2]=(__bf16)(f0.z*scale); r[3]=(__bf16)(f0.w*scale);
                r[4]=(__bf16)(f1.x*scale); r[5]=(__bf16)(f1.y*scale);
                r[6]=(__bf16)(f1.z*scale); r[7]=(__bf16)(f1.w*scale);
                if (fr == 0) a0.h[g] = r; else a1.h[g] = r;
            }
        }
    }

    float mrow[8], lrow[8];
    v8f o[4] = {v8f{}, v8f{}, v8f{}, v8f{}};
    #pragma unroll
    for (int j = 0; j < 8; ++j) { mrow[j] = -1e30f; lrow[j] = 0.f; }

    for (int s0 = 0; s0 < TDIM; s0 += 64) {
        // ---- load K tile [key][d], V tile transposed [d][key] ----
        {
            int key = tid >> 1, dbase = (tid & 1) * 32;
            const float* kp = qkv + (size_t)(b * TDIM + s0 + key) * (3 * CDIM)
                              + CDIM + h * HD + dbase;
            const float* vp = kp + CDIM;
            #pragma unroll
            for (int i = 0; i < 8; ++i) {
                float4 f = *(const float4*)(kp + i * 4);
                int d = dbase + i * 4;
                Kt[key][d+0]=(__bf16)f.x; Kt[key][d+1]=(__bf16)f.y;
                Kt[key][d+2]=(__bf16)f.z; Kt[key][d+3]=(__bf16)f.w;
            }
            #pragma unroll
            for (int i = 0; i < 8; ++i) {
                float4 f = *(const float4*)(vp + i * 4);
                int d = dbase + i * 4;
                Vt[d+0][key]=(__bf16)f.x; Vt[d+1][key]=(__bf16)f.y;
                Vt[d+2][key]=(__bf16)f.z; Vt[d+3][key]=(__bf16)f.w;
            }
        }
        __syncthreads();

        #pragma unroll
        for (int kc = 0; kc < 64; kc += 32) {       // 32-key chunks
            // ---- S fragments: c0 = keys kc..kc+15, c1 = kc+16..kc+31 ----
            v8f c0 = v8f{}, c1 = v8f{};
            {
                FragA b0, b1, b2, b3;
                int key0 = kc + l16;
                int key1 = kc + 16 + l16;
                b0.h[0] = *(const bf16x8*)&Kt[key0][half * 16];
                b0.h[1] = *(const bf16x8*)&Kt[key0][half * 16 + 8];
                b1.h[0] = *(const bf16x8*)&Kt[key0][32 + half * 16];
                b1.h[1] = *(const bf16x8*)&Kt[key0][32 + half * 16 + 8];
                b2.h[0] = *(const bf16x8*)&Kt[key1][half * 16];
                b2.h[1] = *(const bf16x8*)&Kt[key1][half * 16 + 8];
                b3.h[0] = *(const bf16x8*)&Kt[key1][32 + half * 16];
                b3.h[1] = *(const bf16x8*)&Kt[key1][32 + half * 16 + 8];
                c0 = wmma_bf16(a0, b0, c0);
                c0 = wmma_bf16(a1, b1, c0);
                c1 = wmma_bf16(a0, b2, c1);
                c1 = wmma_bf16(a1, b3, c1);
            }
            // ---- online softmax: rows live in 16-lane halves ----
            float p0[8], p1[8];
            #pragma unroll
            for (int j = 0; j < 8; ++j) {
                float s0v = c0[j], s1v = c1[j];
                float mloc = fmaxf(s0v, s1v);
                #pragma unroll
                for (int off = 8; off > 0; off >>= 1)
                    mloc = fmaxf(mloc, __shfl_xor(mloc, off, 32));
                float mnew  = fmaxf(mrow[j], mloc);
                float alpha = __expf(mrow[j] - mnew);
                float e0 = __expf(s0v - mnew), e1 = __expf(s1v - mnew);
                float ls = e0 + e1;
                #pragma unroll
                for (int off = 8; off > 0; off >>= 1)
                    ls += __shfl_xor(ls, off, 32);
                lrow[j] = lrow[j] * alpha + ls;
                mrow[j] = mnew;
                p0[j] = e0; p1[j] = e1;
                #pragma unroll
                for (int f = 0; f < 4; ++f) o[f][j] *= alpha;
            }
            // ---- P (C-layout) -> LDS -> A-layout fragment ----
            #pragma unroll
            for (int j = 0; j < 8; ++j) {
                Ps[wave][j + 8 * half][l16]      = (__bf16)p0[j];
                Ps[wave][j + 8 * half][16 + l16] = (__bf16)p1[j];
            }
            __syncthreads();
            FragA pA;
            {
                int klo = half * 16;
                pA.h[0] = *(const bf16x8*)&Ps[wave][l16][klo];
                pA.h[1] = *(const bf16x8*)&Ps[wave][l16][klo + 8];
            }
            // ---- O += P V : B frag K = keys kc+half*16+0..15, N = d ----
            FragA bv[4];
            #pragma unroll
            for (int f = 0; f < 4; ++f) {
                int d = f * 16 + l16;
                bv[f].h[0] = *(const bf16x8*)&Vt[d][kc + half * 16];
                bv[f].h[1] = *(const bf16x8*)&Vt[d][kc + half * 16 + 8];
            }
            #pragma unroll
            for (int f = 0; f < 4; ++f)
                o[f] = wmma_bf16(pA, bv[f], o[f]);
            __syncthreads();
        }
    }

    // ---- normalize and write y (bf16) ----
    int trow_base = b * TDIM + qt * 64 + wave * 16;
    #pragma unroll
    for (int f = 0; f < 4; ++f) {
        #pragma unroll
        for (int j = 0; j < 8; ++j) {
            int t = trow_base + j + 8 * half;
            float val = o[f][j] / lrow[j];
            y[(size_t)t * CDIM + h * HD + f * 16 + l16] = (__bf16)val;
        }
    }
}

// =====================================================================
// Loss: log-softmax NLL with ignore_index=0 mask
// =====================================================================
__global__ void loss_zero(float* acc) { if (threadIdx.x < 2) acc[threadIdx.x] = 0.f; }

__global__ __launch_bounds__(256)
void loss_row(const float* __restrict__ logits, const int* __restrict__ targets,
              float* __restrict__ acc) {
    __shared__ float red[256];
    int row = blockIdx.x, tid = threadIdx.x;
    const float* lr = logits + (size_t)row * VDIM;

    float mx = -1e30f;
    for (int c = tid; c < VDIM; c += 256) mx = fmaxf(mx, lr[c]);
    red[tid] = mx; __syncthreads();
    for (int o = 128; o > 0; o >>= 1) { if (tid < o) red[tid] = fmaxf(red[tid], red[tid + o]); __syncthreads(); }
    float M = red[0]; __syncthreads();

    float s = 0.f;
    for (int c = tid; c < VDIM; c += 256) s += __expf(lr[c] - M);
    red[tid] = s; __syncthreads();
    for (int o = 128; o > 0; o >>= 1) { if (tid < o) red[tid] += red[tid + o]; __syncthreads(); }
    float S = red[0];

    if (tid == 0) {
        int tgt = targets[row];
        if (tgt != 0) {
            float nll = logf(S) + M - lr[tgt];
            atomicAdd(&acc[0], nll);
            atomicAdd(&acc[1], 1.f);
        }
    }
}

__global__ void loss_final(const float* __restrict__ acc, float* __restrict__ out) {
    if (threadIdx.x == 0) out[0] = acc[0] / fmaxf(acc[1], 1.f);
}

// =====================================================================
// Host orchestration
// =====================================================================
extern "C" void kernel_launch(void* const* d_in, const int* in_sizes, int n_in,
                              void* d_out, int out_size, void* d_ws, size_t ws_size,
                              hipStream_t stream) {
    const int*   idx     = (const int*)  d_in[0];
    const int*   targets = (const int*)  d_in[1];
    const float* wte     = (const float*)d_in[2];
    const float* wpe     = (const float*)d_in[3];
    const float* ln1_g   = (const float*)d_in[4];
    const float* ln1_b   = (const float*)d_in[5];
    const float* w_qkv   = (const float*)d_in[6];
    const float* b_qkv   = (const float*)d_in[7];
    const float* w_proj  = (const float*)d_in[8];
    const float* b_proj  = (const float*)d_in[9];
    const float* ln2_g   = (const float*)d_in[10];
    const float* ln2_b   = (const float*)d_in[11];
    const float* w_fc    = (const float*)d_in[12];
    const float* b_fc    = (const float*)d_in[13];
    const float* w_fc2   = (const float*)d_in[14];
    const float* b_fc2   = (const float*)d_in[15];
    const float* lnf_g   = (const float*)d_in[16];
    const float* lnf_b   = (const float*)d_in[17];
    const float* lm_w    = (const float*)d_in[18];

    const int M = BDIM * TDIM;                     // 4096 rows

    // workspace layout (all 256B aligned)
    char* w = (char*)d_ws;
    float*  x    = (float*) w; w += (size_t)M * CDIM * 4;          // residual
    float*  qkv  = (float*) w; w += (size_t)M * 3 * CDIM * 4;      // qkv fp32
    __bf16* hbf  = (__bf16*)w; w += (size_t)M * CDIM * 2;          // LN out bf16
    __bf16* ybf  = (__bf16*)w; w += (size_t)M * CDIM * 2;          // attn out bf16
    __bf16* mbf  = (__bf16*)w; w += (size_t)M * 4 * CDIM * 2;      // gelu out bf16
    float*  acc  = (float*) w;                                     // loss accum

    embed_kernel<<<M, 256, 0, stream>>>(idx, wte, wpe, x);

    for (int l = 0; l < LNUM; ++l) {
        layernorm_bf16<<<M, 256, 0, stream>>>(x, ln1_g + l * CDIM, ln1_b + l * CDIM, hbf);
        gemm_wmma<<<dim3(3 * CDIM / 64, M / 64), 128, 0, stream>>>(
            hbf, w_qkv + (size_t)l * CDIM * 3 * CDIM, b_qkv + (size_t)l * 3 * CDIM,
            nullptr, qkv, nullptr, M, 3 * CDIM, CDIM, 0, 0);
        attn_flash<<<BDIM * HNUM * (TDIM / 64), 128, 0, stream>>>(qkv, ybf);
        gemm_wmma<<<dim3(CDIM / 64, M / 64), 128, 0, stream>>>(
            ybf, w_proj + (size_t)l * CDIM * CDIM, b_proj + (size_t)l * CDIM,
            x, x, nullptr, M, CDIM, CDIM, 0, 0);
        layernorm_bf16<<<M, 256, 0, stream>>>(x, ln2_g + l * CDIM, ln2_b + l * CDIM, hbf);
        gemm_wmma<<<dim3(4 * CDIM / 64, M / 64), 128, 0, stream>>>(
            hbf, w_fc + (size_t)l * CDIM * 4 * CDIM, b_fc + (size_t)l * 4 * CDIM,
            nullptr, nullptr, mbf, M, 4 * CDIM, CDIM, 0, 1);
        gemm_wmma<<<dim3(CDIM / 64, M / 64), 128, 0, stream>>>(
            mbf, w_fc2 + (size_t)l * 4 * CDIM * CDIM, b_fc2 + (size_t)l * CDIM,
            x, x, nullptr, M, CDIM, 4 * CDIM, 0, 0);
    }

    layernorm_bf16<<<M, 256, 0, stream>>>(x, lnf_g, lnf_b, hbf);

    float* logits = (float*)d_out;
    gemm_wmma<<<dim3((VDIM + 63) / 64, M / 64), 128, 0, stream>>>(
        hbf, lm_w, nullptr, nullptr, logits, nullptr, M, VDIM, CDIM, 1, 0);

    loss_zero<<<1, 32, 0, stream>>>(acc);
    loss_row<<<M, 256, 0, stream>>>(logits, targets, acc);
    loss_final<<<1, 32, 0, stream>>>(acc, logits + (size_t)M * VDIM);
}
